// CorrelationLayer_9363028706363
// MI455X (gfx1250) — matrix operand order repaired and checked by reading, not compiled, not run on hardware
//
#include <hip/hip_runtime.h>
#include <hip/hip_bf16.h>

typedef __attribute__((ext_vector_type(16))) __bf16 v16bf;
typedef __attribute__((ext_vector_type(8)))  float  v8f;

#define B_DIM 8
#define C_DIM 128
#define H_DIM 128
#define W_DIM 192
#define NDISP 80   // 9*9 - 1 displacements

// fp32 -> bf16, round-to-nearest-even
__device__ __forceinline__ unsigned short f32_to_bf16(float f) {
    unsigned int u = __float_as_uint(f);
    unsigned int r = u + 0x7FFFu + ((u >> 16) & 1u);
    return (unsigned short)(r >> 16);
}

// Workgroup: 9 waves (288 threads). Wave k handles row-shift i = k-4.
// Per workgroup: output block (b, h, w0..w0+15) for all 80 displacements.
__global__ __launch_bounds__(288)
void corr_wmma_bf16_kernel(const float* __restrict__ x1,
                           const float* __restrict__ x2,
                           float* __restrict__ out) {
    // A-operand image: [chunk(4)][lane(32)][half(16)] bf16  (exact WMMA A layout)
    __shared__ __align__(32) unsigned short ldsA[4 * 32 * 16];           // 4 KB
    // x2 strip: [i(9)][chunk(4)][w(24)][c(32)] bf16, channel-contiguous
    __shared__ __align__(32) unsigned short ldsB[9 * 4 * 24 * 32];       // 54 KB

    const int wb = blockIdx.x;      // 0..11  (W/16)
    const int h  = blockIdx.y;      // 0..127
    const int b  = blockIdx.z;      // 0..7
    const int w0 = wb * 16;
    const int t  = threadIdx.x;     // 0..287

    // ---- Fill x1 A-image (coalesced over m; slot computed by inverse layout map) ----
    // A 16x32 layout (contiguous K, matching B): lane = m + 16*khalf; half j = k & 15,
    // khalf = k >> 4.
    for (int s = t; s < 2048; s += 288) {
        int m = s & 15;
        int c = s >> 4;                       // 0..127
        int chunk = c >> 5;
        int kin = c & 31;
        int khalf = kin >> 4;                 // {16..31} -> 1
        int rem = kin - 8 * khalf;            // in {0..23}
        int v = (rem < 8) ? (rem >> 1) : (4 + ((rem - 16) >> 1));
        int j = kin & 15;
        int lane = m + 16 * khalf;
        float val = x1[(((size_t)b * C_DIM + c) * H_DIM + h) * W_DIM + (w0 + m)];
        ldsA[chunk * 512 + lane * 16 + j] = f32_to_bf16(val);
    }

    // ---- Fill x2 strip: rows h-4..h+4, cols w0-4..w0+19, zero-padded ----
    for (int s = t; s < 9 * 128 * 24; s += 288) {
        int w = s % 24;
        int c = (s / 24) % 128;
        int i = s / (24 * 128);
        int hrow = h + i - 4;
        int col = w0 - 4 + w;
        float val = 0.0f;
        if ((unsigned)hrow < (unsigned)H_DIM && (unsigned)col < (unsigned)W_DIM)
            val = x2[(((size_t)b * C_DIM + c) * H_DIM + hrow) * W_DIM + col];
        int chunk = c >> 5, cin = c & 31;
        ldsB[(((i * 4 + chunk) * 24) + w) * 32 + cin] = f32_to_bf16(val);
    }
    __syncthreads();

    const int wave = t >> 5;        // 0..8 : row shift index (i = wave-4)
    const int lane = t & 31;
    const int n    = lane & 15;     // Gram column (window position)
    const int kblk = lane >> 4;     // K half-block for B operand

    v8f acc0 = {0.f,0.f,0.f,0.f,0.f,0.f,0.f,0.f};   // off = -4 Gram
    v8f acc1 = {0.f,0.f,0.f,0.f,0.f,0.f,0.f,0.f};   // off = +4 Gram

    #pragma unroll
    for (int chunk = 0; chunk < 4; ++chunk) {
        v16bf a = *reinterpret_cast<const v16bf*>(ldsA + chunk * 512 + lane * 16);
        const unsigned short* Brow = ldsB + ((wave * 4 + chunk) * 24) * 32;
        // B layout (16-bit, 32x16): lane -> (n, kblk); half j <-> K = 16*kblk + j
        v16bf b0 = *reinterpret_cast<const v16bf*>(Brow + (n + 0) * 32 + kblk * 16); // cols w0-4+n
        v16bf b1 = *reinterpret_cast<const v16bf*>(Brow + (n + 8) * 32 + kblk * 16); // cols w0+4+n
        acc0 = __builtin_amdgcn_wmma_f32_16x16x32_bf16(false, a, false, b0, (short)0, acc0, false, false);
        acc1 = __builtin_amdgcn_wmma_f32_16x16x32_bf16(false, a, false, b1, (short)0, acc1, false, false);
    }

    // ---- Banded scatter-store. Each (m,j) comes from exactly one Gram/lane. ----
    const float inv = 0.08838834764831845f;   // 1/sqrt(128)
    const int i  = wave - 4;
    const int mh = lane >> 4;                 // C/D layout: VGPR r -> M = r + 8*mh

    #pragma unroll
    for (int r = 0; r < 8; ++r) {
        int m = r + 8 * mh;
        // off = -4 tile: j = n - m - 4 ; covers m+j <= 11
        {
            int j = n - m - 4;
            if (j >= -4 && j <= 4 && !(i == 0 && j == 0)) {
                int idx = (i + 4) * 9 + (j + 4);
                int d = idx - (idx > 40 ? 1 : 0);
                out[(((size_t)b * NDISP + d) * H_DIM + h) * W_DIM + (w0 + m)] = acc0[r] * inv;
            }
        }
        // off = +4 tile: j = n - m + 4 ; covers m+j >= 12 (enforced by n >= 8)
        {
            int j = n - m + 4;
            if (n >= 8 && j >= -4 && j <= 4 && !(i == 0 && j == 0)) {
                int idx = (i + 4) * 9 + (j + 4);
                int d = idx - (idx > 40 ? 1 : 0);
                out[(((size_t)b * NDISP + d) * H_DIM + h) * W_DIM + (w0 + m)] = acc1[r] * inv;
            }
        }
    }
}

extern "C" void kernel_launch(void* const* d_in, const int* in_sizes, int n_in,
                              void* d_out, int out_size, void* d_ws, size_t ws_size,
                              hipStream_t stream) {
    (void)in_sizes; (void)n_in; (void)out_size; (void)d_ws; (void)ws_size;
    const float* x1 = (const float*)d_in[0];
    const float* x2 = (const float*)d_in[1];
    float* out = (float*)d_out;

    dim3 grid(W_DIM / 16, H_DIM, B_DIM);   // 12 x 128 x 8 workgroups
    dim3 block(288);                        // 9 wave32's: one per row shift i
    corr_wmma_bf16_kernel<<<grid, block, 0, stream>>>(x1, x2, out);
}
